// SparseCrossAttentionLayer_60945585930256
// MI455X (gfx1250) — matrix operand order repaired and checked by reading, not compiled, not run on hardware
//
#include <hip/hip_runtime.h>
#include <math.h>

// ---------------- problem constants ----------------
#define NN    10000
#define DD    256
#define HH    8
#define DHH   32
#define EE    160000
#define NEIGH 50
#define CAP   192            // per-node bucket capacity (avg deg ~33, safe margin)
#define QKSCALE 0.17677669529663687f   // 1/sqrt(32)

typedef _Float16 half_t;
typedef __attribute__((ext_vector_type(16))) _Float16 v16h;
typedef __attribute__((ext_vector_type(8)))  _Float16 v8h;
typedef __attribute__((ext_vector_type(8)))  float    v8f;

// ---------------- wave32 reductions ----------------
__device__ __forceinline__ float wave_sum(float x) {
#pragma unroll
  for (int o = 16; o > 0; o >>= 1) x += __shfl_xor(x, o, 32);
  return x;
}
__device__ __forceinline__ float wave_max(float x) {
#pragma unroll
  for (int o = 16; o > 0; o >>= 1) x = fmaxf(x, __shfl_xor(x, o, 32));
  return x;
}

// ---------------- kernel 0: zero int buffer ----------------
__global__ void k_zero_i32(int* p, int n) {
  int i = blockIdx.x * blockDim.x + threadIdx.x;
  if (i < n) p[i] = 0;
}

// ---------------- kernel 1: pack weights into WMMA B-fragment order -----
// B tile (kt,jt) is 32x16 (KxN). Lane l holds column c=l%16, K = (l/16)*16 + h,
// h = 0..15 contiguous halves -> GEMM loads are two 16B vector loads per lane.
__global__ void k_pack_w(const float* __restrict__ Wq, const float* __restrict__ Wk,
                         const float* __restrict__ Wv, const float* __restrict__ Wo,
                         half_t* __restrict__ wpack) {
  int t = blockIdx.x * blockDim.x + threadIdx.x;      // 0..65535
  int w = blockIdx.y;                                 // which weight
  const float* W = (w == 0) ? Wq : (w == 1) ? Wk : (w == 2) ? Wv : Wo;
  int h  = t & 15;
  int l  = (t >> 4) & 31;
  int jt = (t >> 9) & 15;
  int kt = t >> 13;
  int j = jt * 16 + (l & 15);
  int k = kt * 32 + (l >> 4) * 16 + h;
  // q = x @ W^T  =>  B[k][j] = W[j][k]
  wpack[(size_t)w * 65536 + t] = (half_t)W[j * DD + k];
}

// ---------------- kernel 2: pos features + residual + LayerNorm ---------
// one wave per row; lane owns 8 elements strided by 32
__global__ void k_pos_ln(const float* __restrict__ query, const float* __restrict__ kv,
                         const float* __restrict__ deg,   const float* __restrict__ cc,
                         const float* __restrict__ pe,
                         const float* __restrict__ Wc, const float* __restrict__ bc,
                         const float* __restrict__ Wd, const float* __restrict__ bd,
                         const float* __restrict__ g_q,  const float* __restrict__ b_q,
                         const float* __restrict__ g_kv, const float* __restrict__ b_kv,
                         half_t* __restrict__ qn, half_t* __restrict__ kvn,
                         float* __restrict__ xqres) {
  int row  = blockIdx.x * 8 + (threadIdx.x >> 5);
  int lane = threadIdx.x & 31;
  if (row >= NN) return;
  float xq[8], xk[8];
  float sq = 0.f, sq2 = 0.f, sk = 0.f, sk2 = 0.f;
  float ccv = cc[row], dgv = deg[row];
#pragma unroll
  for (int i = 0; i < 8; ++i) {
    int d = lane + i * 32;
    float pos;
    if (d < 128)      pos = pe[(size_t)row * DD + d];
    else if (d < 192) pos = ccv * Wc[d - 128] + bc[d - 128];
    else              pos = dgv * Wd[d - 192] + bd[d - 192];
    float a = query[(size_t)row * DD + d] + pos;
    float b = kv[(size_t)row * DD + d] + pos;
    xq[i] = a; xk[i] = b;
    sq += a; sq2 += a * a; sk += b; sk2 += b * b;
  }
  sq = wave_sum(sq);  sq2 = wave_sum(sq2);
  sk = wave_sum(sk);  sk2 = wave_sum(sk2);
  float mq = sq * (1.f / DD), mk = sk * (1.f / DD);
  float rq = rsqrtf(sq2 * (1.f / DD) - mq * mq + 1e-5f);
  float rk = rsqrtf(sk2 * (1.f / DD) - mk * mk + 1e-5f);
#pragma unroll
  for (int i = 0; i < 8; ++i) {
    int d = lane + i * 32;
    xqres[(size_t)row * DD + d] = xq[i];
    qn [(size_t)row * DD + d] = (half_t)((xq[i] - mq) * rq * g_q[d]  + b_q[d]);
    kvn[(size_t)row * DD + d] = (half_t)((xk[i] - mk) * rk * g_kv[d] + b_kv[d]);
  }
}

// ---------------- kernel 3: WMMA GEMM  Out = A(f16) x Bpack(f16) + bias --
// one wave computes one 16x16 tile, K=256 in 8 steps of 32
__global__ __launch_bounds__(256) void k_gemm_wmma(const half_t* __restrict__ A,
                                                   const half_t* __restrict__ Bp,
                                                   const float* __restrict__ bias,
                                                   float* __restrict__ Out) {
  int rt   = blockIdx.x;                       // row tile (0..624)
  int wave = threadIdx.x >> 5;
  int jt   = blockIdx.y * 8 + wave;            // col tile (0..15)
  int lane = threadIdx.x & 31;
  int g    = lane >> 4;                        // half-lane group
  int m    = lane & 15;                        // row within tile (A side)
  int row  = rt * 16 + m;

  v8f acc = {};
#pragma unroll
  for (int kt = 0; kt < 8; ++kt) {
    // A fragment: lane holds K = kt*32 + {g*16..g*16+15}
    const half_t* ar = A + (size_t)row * DD + kt * 32 + g * 16;
    v8h alo = *reinterpret_cast<const v8h*>(ar);
    v8h ahi = *reinterpret_cast<const v8h*>(ar + 8);
    v16h a;
#pragma unroll
    for (int i = 0; i < 8; ++i) { a[i] = alo[i]; a[i + 8] = ahi[i]; }
    // B fragment: pre-packed contiguous 16 halves per lane
    const half_t* br = Bp + (((size_t)kt * 16 + jt) * 32 + lane) * 16;
    v8h blo = *reinterpret_cast<const v8h*>(br);
    v8h bhi = *reinterpret_cast<const v8h*>(br + 8);
    v16h b;
#pragma unroll
    for (int i = 0; i < 8; ++i) { b[i] = blo[i]; b[i + 8] = bhi[i]; }
    acc = __builtin_amdgcn_wmma_f32_16x16x32_f16(false, a, false, b,
                                                 (short)0, acc, false, false);
  }
  // D layout: VGPR i -> row i + 8*g, col = lane%16
  int j = jt * 16 + (lane & 15);
  float bj = bias[j];
#pragma unroll
  for (int i = 0; i < 8; ++i) {
    int r = rt * 16 + i + 8 * g;
    Out[(size_t)r * DD + j] = acc[i] + bj;
  }
}

// ---------------- kernel 4: scatter edges into per-node buckets ----------
// key = original concatenated index (unique) -> gives jnp stable-sort order
__global__ void k_scatter(const int* __restrict__ ei, int* __restrict__ cnt,
                          int* __restrict__ bkey, int* __restrict__ bnbr) {
  int t = blockIdx.x * blockDim.x + threadIdx.x;
  if (t < EE) {
    int r = ei[t], c = ei[EE + t];
    int s = atomicAdd(&cnt[r], 1);
    if (s < CAP) { bkey[(size_t)r * CAP + s] = t;      bnbr[(size_t)r * CAP + s] = c; }
    s = atomicAdd(&cnt[c], 1);
    if (s < CAP) { bkey[(size_t)c * CAP + s] = EE + t; bnbr[(size_t)c * CAP + s] = r; }
  } else if (t < EE + NN) {
    int n = t - EE;
    int s = atomicAdd(&cnt[n], 1);
    if (s < CAP) { bkey[(size_t)n * CAP + s] = 2 * EE + n; bnbr[(size_t)n * CAP + s] = n; }
  }
}

// ---------------- kernel 5: select 50 smallest-key neighbors per node ----
__global__ void k_select(const int* __restrict__ cnt, const int* __restrict__ bkey,
                         const int* __restrict__ bnbr, int* __restrict__ sel,
                         int* __restrict__ selcnt) {
  __shared__ int sk[CAP], sn[CAP];
  int n = blockIdx.x, tid = threadIdx.x;
  int c = cnt[n]; if (c > CAP) c = CAP;
  for (int i = tid; i < c; i += blockDim.x) {
    sk[i] = bkey[(size_t)n * CAP + i];
    sn[i] = bnbr[(size_t)n * CAP + i];
  }
  __syncthreads();
  for (int i = tid; i < c; i += blockDim.x) {
    int ki = sk[i], rank = 0;
    for (int j = 0; j < c; ++j) rank += (sk[j] < ki);   // keys unique
    if (rank < NEIGH) sel[(size_t)n * NEIGH + rank] = sn[i];
  }
  if (tid == 0) selcnt[n] = (c < NEIGH) ? c : NEIGH;
}

// ---------------- kernel 6: per-node sparse attention --------------------
// block = 256 (8 waves, one wave per head); deterministic fixed-order sums
__global__ __launch_bounds__(256) void k_attn(const float* __restrict__ qf,
                                              const float* __restrict__ kf,
                                              const float* __restrict__ vf,
                                              const int* __restrict__ sel,
                                              const int* __restrict__ selcnt,
                                              half_t* __restrict__ outh) {
  __shared__ float lq[DD];
  __shared__ int   ls[NEIGH];
  __shared__ int   lc;
  __shared__ float lat[HH][NEIGH + 2];
  int n = blockIdx.x, tid = threadIdx.x;
  lq[tid] = qf[(size_t)n * DD + tid];
  if (tid == 0) lc = selcnt[n];
  if (tid < NEIGH) ls[tid] = sel[(size_t)n * NEIGH + tid];
  __syncthreads();
  int cntv = lc;
  int h = tid >> 5, lane = tid & 31;
  int qb = h * DHH;

  float s0 = -1e30f, s1 = -1e30f;
  if (lane < cntv) {
    const float* kr = kf + (size_t)ls[lane] * DD + qb;
    float s = 0.f;
#pragma unroll
    for (int d = 0; d < DHH; d += 4) {
      float4 k4 = *reinterpret_cast<const float4*>(kr + d);
      s += lq[qb + d] * k4.x + lq[qb + d + 1] * k4.y +
           lq[qb + d + 2] * k4.z + lq[qb + d + 3] * k4.w;
    }
    s0 = s * QKSCALE;
  }
  if (32 + lane < cntv) {
    const float* kr = kf + (size_t)ls[32 + lane] * DD + qb;
    float s = 0.f;
#pragma unroll
    for (int d = 0; d < DHH; d += 4) {
      float4 k4 = *reinterpret_cast<const float4*>(kr + d);
      s += lq[qb + d] * k4.x + lq[qb + d + 1] * k4.y +
           lq[qb + d + 2] * k4.z + lq[qb + d + 3] * k4.w;
    }
    s1 = s * QKSCALE;
  }
  float mx = wave_max(fmaxf(s0, s1));
  float e0 = (lane < cntv)      ? expf(s0 - mx) : 0.f;
  float e1 = (32 + lane < cntv) ? expf(s1 - mx) : 0.f;
  float inv = 1.f / wave_sum(e0 + e1);
  if (lane < cntv)      lat[h][lane]      = e0 * inv;
  if (32 + lane < cntv) lat[h][32 + lane] = e1 * inv;
  __syncthreads();

  float acc = 0.f;
  for (int r = 0; r < cntv; ++r)
    acc += lat[h][r] * vf[(size_t)ls[r] * DD + qb + lane];
  outh[(size_t)n * DD + qb + lane] = (half_t)acc;
}

// ---------------- kernel 7: final residual + LayerNorm -------------------
__global__ void k_out_ln(const float* __restrict__ proj, const float* __restrict__ xqres,
                         const float* __restrict__ g, const float* __restrict__ b,
                         float* __restrict__ out) {
  int row  = blockIdx.x * 8 + (threadIdx.x >> 5);
  int lane = threadIdx.x & 31;
  if (row >= NN) return;
  float x[8]; float s = 0.f, s2 = 0.f;
#pragma unroll
  for (int i = 0; i < 8; ++i) {
    int d = lane + i * 32;
    float v = proj[(size_t)row * DD + d] + xqres[(size_t)row * DD + d];
    x[i] = v; s += v; s2 += v * v;
  }
  s = wave_sum(s); s2 = wave_sum(s2);
  float m = s * (1.f / DD);
  float r = rsqrtf(s2 * (1.f / DD) - m * m + 1e-5f);
#pragma unroll
  for (int i = 0; i < 8; ++i) {
    int d = lane + i * 32;
    out[(size_t)row * DD + d] = (x[i] - m) * r * g[d] + b[d];
  }
}

// ---------------- host side ------------------------------------------------
extern "C" void kernel_launch(void* const* d_in, const int* in_sizes, int n_in,
                              void* d_out, int out_size, void* d_ws, size_t ws_size,
                              hipStream_t stream) {
  const float* query = (const float*)d_in[0];
  const float* keyv  = (const float*)d_in[1];
  const float* ndeg  = (const float*)d_in[2];
  const float* ccf   = (const float*)d_in[3];
  const float* pe    = (const float*)d_in[4];
  const float* Wc    = (const float*)d_in[5];
  const float* bc    = (const float*)d_in[6];
  const float* Wd    = (const float*)d_in[7];
  const float* bd    = (const float*)d_in[8];
  const float* g_q   = (const float*)d_in[9];
  const float* b_q   = (const float*)d_in[10];
  const float* g_kv  = (const float*)d_in[11];
  const float* b_kv  = (const float*)d_in[12];
  const float* g_out = (const float*)d_in[13];
  const float* b_out = (const float*)d_in[14];
  const float* Wq    = (const float*)d_in[15];
  const float* bq    = (const float*)d_in[16];
  const float* Wk    = (const float*)d_in[17];
  const float* bk    = (const float*)d_in[18];
  const float* Wv    = (const float*)d_in[19];
  const float* bv    = (const float*)d_in[20];
  const float* Wo    = (const float*)d_in[21];
  const float* bo    = (const float*)d_in[22];
  const int*   ei    = (const int*)d_in[23];
  float* out = (float*)d_out;

  // workspace carve-out (256B aligned)
  char* ws = (char*)d_ws;
  size_t off = 0;
  auto carve = [&](size_t bytes) -> char* {
    char* p = ws + off;
    off = (off + bytes + 255) & ~(size_t)255;
    return p;
  };
  half_t* qn    = (half_t*)carve((size_t)NN * DD * 2);
  half_t* kvn   = (half_t*)carve((size_t)NN * DD * 2);
  float*  xqres = (float*) carve((size_t)NN * DD * 4);
  float*  qf    = (float*) carve((size_t)NN * DD * 4);
  float*  kf    = (float*) carve((size_t)NN * DD * 4);
  float*  vf    = (float*) carve((size_t)NN * DD * 4);
  half_t* atth  = (half_t*)carve((size_t)NN * DD * 2);
  float*  proj  = (float*) carve((size_t)NN * DD * 4);
  half_t* wpack = (half_t*)carve((size_t)4 * 65536 * 2);
  int*    cnt   = (int*)   carve((size_t)NN * 4);
  int*    bkey  = (int*)   carve((size_t)NN * CAP * 4);
  int*    bnbr  = (int*)   carve((size_t)NN * CAP * 4);
  int*    sel   = (int*)   carve((size_t)NN * NEIGH * 4);
  int*    selcnt= (int*)   carve((size_t)NN * 4);
  (void)ws_size; (void)n_in; (void)in_sizes; (void)out_size;

  // 0) zero bucket counters
  k_zero_i32<<<(NN + 255) / 256, 256, 0, stream>>>(cnt, NN);
  // 1) pack weights into WMMA B-fragment layout (f16)
  k_pack_w<<<dim3(256, 4), 256, 0, stream>>>(Wq, Wk, Wv, Wo, wpack);
  // 2) positional features + residual stash + LN (f16 outputs for WMMA)
  k_pos_ln<<<(NN + 7) / 8, 256, 0, stream>>>(query, keyv, ndeg, ccf, pe,
                                             Wc, bc, Wd, bd, g_q, b_q, g_kv, b_kv,
                                             qn, kvn, xqres);
  // 3) edge bucket scatter + top-50 stable selection
  k_scatter<<<(EE + NN + 255) / 256, 256, 0, stream>>>(ei, cnt, bkey, bnbr);
  k_select <<<NN, 64, 0, stream>>>(cnt, bkey, bnbr, sel, selcnt);
  // 4) Q/K/V projections on the matrix pipes
  k_gemm_wmma<<<dim3(NN / 16, 2), 256, 0, stream>>>(qn,  wpack + 0 * 65536, bq, qf);
  k_gemm_wmma<<<dim3(NN / 16, 2), 256, 0, stream>>>(kvn, wpack + 1 * 65536, bk, kf);
  k_gemm_wmma<<<dim3(NN / 16, 2), 256, 0, stream>>>(kvn, wpack + 2 * 65536, bv, vf);
  // 5) sparse segment-softmax attention (L2-resident gathers)
  k_attn<<<NN, 256, 0, stream>>>(qf, kf, vf, sel, selcnt, atth);
  // 6) output projection + residual + LN
  k_gemm_wmma<<<dim3(NN / 16, 2), 256, 0, stream>>>(atth, wpack + 3 * 65536, bo, proj);
  k_out_ln<<<(NN + 7) / 8, 256, 0, stream>>>(proj, xqres, g_out, b_out, out);
}
